// FractalDiffusionModel_62766652063772
// MI455X (gfx1250) — matrix operand order repaired and checked by reading, not compiled, not run on hardware
//
#include <hip/hip_runtime.h>
#include <hip/hip_bf16.h>
#include <math.h>

typedef __bf16 bf16_t;
typedef __bf16  v16bf __attribute__((ext_vector_type(16)));
typedef float   v8f   __attribute__((ext_vector_type(8)));

#define WMMA_BF16(a, b, c) \
    __builtin_amdgcn_wmma_f32_16x16x32_bf16(false, (a), false, (b), (short)0, (c), false, false)

union AFragU { v16bf v; unsigned u[8]; };

// ---------------------------------------------------------------------------
// Model dimensions (fixed by the reference)
// ---------------------------------------------------------------------------
static constexpr int Bc  = 16;
static constexpr int Tc  = 512;
static constexpr int Lc  = 12;
static constexpr int Hc  = 16;
static constexpr int Dc  = 1024;
static constexpr int HDc = 64;
static constexpr int V0c = 1025;

// Epilogue modes for the WMMA GEMM
static constexpr int EPI_F32     = 0;
static constexpr int EPI_RESID   = 1;
static constexpr int EPI_BF      = 2;
static constexpr int EPI_GELU_BF = 3;
static constexpr int EPI_HEAD    = 4;

__device__ __forceinline__ float gelu_exact(float v) {
    return 0.5f * v * (1.0f + erff(v * 0.70710678118654752f));
}

// ---------------------------------------------------------------------------
// DPP16 butterfly reductions across a 16-lane half (stays within each half,
// matching the WMMA C-matrix lane split).
// ---------------------------------------------------------------------------
template <int CTRL>
__device__ __forceinline__ float dppf(float x) {
    int xi = __float_as_int(x);
    return __int_as_float(__builtin_amdgcn_update_dpp(xi, xi, CTRL, 0xF, 0xF, true));
}
__device__ __forceinline__ float rowmax16(float x) {
    x = fmaxf(x, dppf<0xB1>(x));    // quad_perm(1,0,3,2)  : xor 1
    x = fmaxf(x, dppf<0x4E>(x));    // quad_perm(2,3,0,1)  : xor 2
    x = fmaxf(x, dppf<0x141>(x));   // row_half_mirror     : xor 7
    x = fmaxf(x, dppf<0x140>(x));   // row_mirror          : xor 15
    return x;
}
__device__ __forceinline__ float rowsum16(float x) {
    x += dppf<0xB1>(x);
    x += dppf<0x4E>(x);
    x += dppf<0x141>(x);
    x += dppf<0x140>(x);
    return x;
}

// ---------------------------------------------------------------------------
// f32 -> bf16 conversion (weights; redone every call — determinism rule)
// ---------------------------------------------------------------------------
__global__ void f32_to_bf16_kernel(const float* __restrict__ in,
                                   bf16_t* __restrict__ out, size_t n) {
    size_t i = blockIdx.x * (size_t)blockDim.x + threadIdx.x;
    size_t stride = (size_t)gridDim.x * blockDim.x;
    for (; i < n; i += stride) out[i] = (bf16_t)in[i];
}

// ---------------------------------------------------------------------------
// Time embedding features: concat(sin(t*e), cos(t*e)) -> (B, D)
// ---------------------------------------------------------------------------
__global__ void timefeat_kernel(const int* __restrict__ t, float* __restrict__ f,
                                int B, int D) {
    int i = blockIdx.x * blockDim.x + threadIdx.x;
    int half = D / 2;
    if (i >= B * half) return;
    int b = i / half, j = i % half;
    float e = expf((float)j * (-logf(10000.0f) / (float)(half - 1)));
    float a = (float)t[b] * e;
    f[(size_t)b * D + j]        = sinf(a);
    f[(size_t)b * D + half + j] = cosf(a);
}

// Small dense MLP layer (M=16 rows — negligible FLOPs)
__global__ void small_mm_kernel(const float* __restrict__ A, const float* __restrict__ W,
                                const float* __restrict__ bias, float* __restrict__ out,
                                int M, int N, int K, int dogelu) {
    int idx = blockIdx.x * blockDim.x + threadIdx.x;
    if (idx >= M * N) return;
    int m = idx / N, n = idx % N;
    float s = bias ? bias[n] : 0.0f;
    for (int k = 0; k < K; ++k) s += A[(size_t)m * K + k] * W[(size_t)k * N + n];
    if (dogelu) s = gelu_exact(s);
    out[idx] = s;
}

// h = tok_emb[x] + level_emb[level] + te[b]
__global__ void embed_kernel(const int* __restrict__ x, const float* __restrict__ tok,
                             const float* __restrict__ lev, const int* __restrict__ level,
                             const float* __restrict__ te, float* __restrict__ h,
                             int BT, int D, int T) {
    size_t i = blockIdx.x * (size_t)blockDim.x + threadIdx.x;
    size_t n = (size_t)BT * D;
    if (i >= n) return;
    int d  = (int)(i % D);
    int bt = (int)(i / D);
    int b  = bt / T;
    int tokid = x[bt];
    int lv = level[0];
    h[i] = tok[(size_t)tokid * D + d] + lev[(size_t)lv * D + d] + te[(size_t)b * D + d];
}

// ---------------------------------------------------------------------------
// LayerNorm (one block per row), bf16 output (feeds WMMA GEMMs)
// ---------------------------------------------------------------------------
__global__ __launch_bounds__(256)
void layernorm_bf16_kernel(const float* __restrict__ X, const float* __restrict__ g,
                           const float* __restrict__ bta, bf16_t* __restrict__ out, int D) {
    __shared__ float red[256];
    int row = blockIdx.x;
    const float* x = X + (size_t)row * D;
    float s = 0.f, s2 = 0.f;
    for (int i = threadIdx.x; i < D; i += 256) { float v = x[i]; s += v; s2 += v * v; }
    red[threadIdx.x] = s; __syncthreads();
    for (int st = 128; st > 0; st >>= 1) {
        if (threadIdx.x < st) red[threadIdx.x] += red[threadIdx.x + st];
        __syncthreads();
    }
    float mean = red[0] / (float)D;
    __syncthreads();
    red[threadIdx.x] = s2; __syncthreads();
    for (int st = 128; st > 0; st >>= 1) {
        if (threadIdx.x < st) red[threadIdx.x] += red[threadIdx.x + st];
        __syncthreads();
    }
    float var = red[0] / (float)D - mean * mean;
    float rstd = rsqrtf(var + 1e-5f);
    for (int i = threadIdx.x; i < D; i += 256) {
        float v = (x[i] - mean) * rstd * g[i] + bta[i];
        out[(size_t)row * D + i] = (bf16_t)v;
    }
}

// ---------------------------------------------------------------------------
// bf16 WMMA GEMM: C[M,N] = A[M,K] * B[K,N]   (A,B bf16 row-major, f32 acc)
// Block tile 128x128, BK=32, 4 waves, wave tile 64x64 (4x4 of 16x16x32 WMMA).
// Ping-pong LDS double buffer; unroll-2 makes buffer indices compile-time.
// ---------------------------------------------------------------------------
template <int EPI>
__global__ __launch_bounds__(128)
void gemm_bf16_kernel(const bf16_t* __restrict__ A, const bf16_t* __restrict__ Bw,
                      float* __restrict__ Cf, bf16_t* __restrict__ Cb,
                      const float* __restrict__ Resid,
                      int M, int N, int K, int Tdim) {
    constexpr int BM = 128, BN = 128, BK = 32, PAD = 8;
    __shared__ alignas(16) bf16_t sA[2][BM][BK + PAD];
    __shared__ alignas(16) bf16_t sB[2][BN][BK + PAD];   // transposed: sB[s][n][k]

    const int tid  = threadIdx.x;
    const int lane = tid & 31;
    const int wave = tid >> 5;
    const int wm = (wave >> 1) * 64;
    const int wn = (wave & 1) * 64;
    const int lh = lane >> 4;
    const int ll = lane & 15;
    const int rowBase = blockIdx.y * BM;
    const int colBase = blockIdx.x * BN;

    auto loadA = [&](int k0, uint4* ra) {
        #pragma unroll
        for (int i = 0; i < 4; ++i) {
            int idx = tid + i * 128;
            int r   = idx >> 2;
            int kq  = (idx & 3) * 8;
            int gr  = rowBase + r;
            ra[i] = make_uint4(0, 0, 0, 0);
            if (gr < M) ra[i] = *(const uint4*)(A + (size_t)gr * K + k0 + kq);
        }
    };
    auto loadB = [&](int k0, uint4* rb) {
        #pragma unroll
        for (int i = 0; i < 4; ++i) {
            int idx = tid + i * 128;
            int kk  = idx >> 4;
            int nq  = (idx & 15) * 8;
            int gc  = colBase + nq;
            if (gc + 7 < N) {
                rb[i] = *(const uint4*)(Bw + (size_t)(k0 + kk) * N + gc);
            } else {
                alignas(16) bf16_t tmp[8];
                #pragma unroll
                for (int j = 0; j < 8; ++j)
                    tmp[j] = (gc + j < N) ? Bw[(size_t)(k0 + kk) * N + gc + j] : (bf16_t)0.0f;
                rb[i] = *(uint4*)tmp;
            }
        }
    };
    auto storeA = [&](int st, const uint4* ra) {
        #pragma unroll
        for (int i = 0; i < 4; ++i) {
            int idx = tid + i * 128;
            int r   = idx >> 2;
            int kq  = (idx & 3) * 8;
            *(uint4*)(&sA[st][r][kq]) = ra[i];
        }
    };
    auto storeB = [&](int st, const uint4* rb) {
        #pragma unroll
        for (int i = 0; i < 4; ++i) {
            int idx = tid + i * 128;
            int kk  = idx >> 4;
            int nq  = (idx & 15) * 8;
            alignas(16) bf16_t tmp[8];
            *(uint4*)tmp = rb[i];
            #pragma unroll
            for (int j = 0; j < 8; ++j) sB[st][nq + j][kk] = tmp[j];
        }
    };

    v8f acc[4][4] = {};
    const int nK = K / BK;

    uint4 ra[4], rb[4];
    loadA(0, ra); loadB(0, rb);
    storeA(0, ra); storeB(0, rb);
    __syncthreads();

    #pragma unroll 2
    for (int kt = 0; kt < nK; ++kt) {
        const int cur = kt & 1, nxt = cur ^ 1;
        const bool more = (kt + 1) < nK;
        if (more) {
            loadA((kt + 1) * BK, ra);
            loadB((kt + 1) * BK, rb);
        }
        if (kt + 2 < nK) {   // L2 prefetch two tiles ahead (global_prefetch_b8)
            __builtin_prefetch(A + (size_t)(rowBase + tid) * K + (kt + 2) * BK, 0, 1);
            __builtin_prefetch(Bw + (size_t)((kt + 2) * BK + (tid & 31)) * N + colBase, 0, 1);
        }

        // fragments (ISA 7.12.2 layouts) + 16 WMMAs per wave
        AFragU a[4], b[4];
        #pragma unroll
        for (int mi = 0; mi < 4; ++mi) {
            int r = wm + mi * 16 + ll;
            #pragma unroll
            for (int e = 0; e < 8; ++e) {
                int kk = ((e < 4) ? 2 * e : 2 * e + 8) + lh * 8;   // A 16x32 layout
                a[mi].u[e] = *(const unsigned*)(&sA[cur][r][kk]);
            }
        }
        #pragma unroll
        for (int ni = 0; ni < 4; ++ni) {
            int c = wn + ni * 16 + ll;
            #pragma unroll
            for (int e = 0; e < 8; ++e) {
                int kk = ((e < 4) ? 2 * e : 2 * e + 8) + lh * 8;   // B 32x16 layout
                b[ni].u[e] = *(const unsigned*)(&sB[cur][c][kk]);
            }
        }
        #pragma unroll
        for (int mi = 0; mi < 4; ++mi)
            #pragma unroll
            for (int ni = 0; ni < 4; ++ni)
                acc[mi][ni] = WMMA_BF16(a[mi].v, b[ni].v, acc[mi][ni]);

        if (more) {
            storeA(nxt, ra);
            storeB(nxt, rb);
        }
        __syncthreads();
    }

    // epilogue (C layout: VGPR r -> row r + lh*8, col = ll)
    #pragma unroll
    for (int mi = 0; mi < 4; ++mi) {
        #pragma unroll
        for (int ni = 0; ni < 4; ++ni) {
            #pragma unroll
            for (int r = 0; r < 8; ++r) {
                int grow = rowBase + wm + mi * 16 + lh * 8 + r;
                int gcol = colBase + wn + ni * 16 + ll;
                if (grow >= M || gcol >= N) continue;
                float v = acc[mi][ni][r];
                size_t off = (size_t)grow * N + gcol;
                if (EPI == EPI_F32) {
                    Cf[off] = v;
                } else if (EPI == EPI_RESID) {
                    Cf[off] = v + Resid[off];
                } else if (EPI == EPI_BF) {
                    Cb[off] = (bf16_t)v;
                } else if (EPI == EPI_GELU_BF) {
                    Cb[off] = (bf16_t)gelu_exact(v);
                } else if (EPI == EPI_HEAD) {
                    int t = grow % Tdim;
                    if (t == 0) continue;
                    size_t orow = (size_t)(grow / Tdim) * (Tdim - 1) + (t - 1);
                    Cf[orow * N + gcol] = v;
                }
            }
        }
    }
}

// ---------------------------------------------------------------------------
// RoPE + head split: qkv[B,T,3,H,HD] bf16 -> q,k,v [B*H, T, HD] bf16
// ---------------------------------------------------------------------------
__global__ void rope_split_kernel(const bf16_t* __restrict__ qkv, bf16_t* __restrict__ q,
                                  bf16_t* __restrict__ k, bf16_t* __restrict__ v,
                                  int B, int T, int H, int HD) {
    int D = H * HD;
    int hp = HD / 2;
    size_t np = (size_t)B * T * H * hp;
    size_t i = blockIdx.x * (size_t)blockDim.x + threadIdx.x;
    if (i >= np) return;
    int j = (int)(i % hp);
    size_t r = i / hp;
    int h = (int)(r % H); r /= H;
    int t = (int)(r % T);
    int b = (int)(r / T);
    float freq = powf(10000.0f, -2.0f * (float)j / (float)HD);
    float ang = (float)t * freq;
    float c = cosf(ang), s = sinf(ang);
    size_t base = ((size_t)(b * T + t) * 3) * D + h * HD + 2 * j;
    float q1 = (float)qkv[base],         q2 = (float)qkv[base + 1];
    float k1 = (float)qkv[base + D],     k2 = (float)qkv[base + D + 1];
    float v1 = (float)qkv[base + 2 * D], v2 = (float)qkv[base + 2 * D + 1];
    size_t ob = ((size_t)(b * H + h) * T + t) * HD + 2 * j;
    q[ob]     = (bf16_t)(q1 * c - q2 * s);
    q[ob + 1] = (bf16_t)(q1 * s + q2 * c);
    k[ob]     = (bf16_t)(k1 * c - k2 * s);
    k[ob + 1] = (bf16_t)(k1 * s + k2 * c);
    v[ob]     = (bf16_t)v1;
    v[ob + 1] = (bf16_t)v2;
}

// ---------------------------------------------------------------------------
// Flash-style attention (full softmax, no causal mask — matches reference).
// Grid: (T/64, B*H). 4 waves; each wave owns 16 query rows. K/V tiles
// double-buffered in LDS; softmax reductions in DPP16; all tile fragments
// hoisted into registers before each WMMA chain so the 8 matrix ops issue
// back-to-back behind a single DS wait.
// ---------------------------------------------------------------------------
__global__ __launch_bounds__(128)
void attn_kernel(const bf16_t* __restrict__ Q, const bf16_t* __restrict__ Km,
                 const bf16_t* __restrict__ Vm, bf16_t* __restrict__ O,
                 int T, int H, int HD, float scale) {
    constexpr int BQ = 64, BKV = 64, PAD = 8;
    __shared__ alignas(16) bf16_t sQ[BQ][64 + PAD];
    __shared__ alignas(16) bf16_t sK[2][BKV][64 + PAD];   // [key][dim] (acts as B^T)
    __shared__ alignas(16) bf16_t sVt[2][64][BKV + PAD];  // [dim][key]
    __shared__ alignas(16) bf16_t sP[BQ][BKV + PAD];      // probs, A-side repack

    const int tid = threadIdx.x, lane = tid & 31, wave = tid >> 5;
    const int lh = lane >> 4, ll = lane & 15;
    const int bh = blockIdx.y;
    const int b = bh / H, h = bh % H;
    const int qBase = blockIdx.x * BQ;
    const int D = H * HD;
    const bf16_t* Qh = Q + (size_t)bh * T * HD;
    const bf16_t* Kh = Km + (size_t)bh * T * HD;
    const bf16_t* Vh = Vm + (size_t)bh * T * HD;

    // stage Q tile + KV tile 0
    #pragma unroll
    for (int i = 0; i < 4; ++i) {
        int idx = tid + i * 128;
        int r = idx >> 3, dq = (idx & 7) * 8;
        *(uint4*)&sQ[r][dq] = *(const uint4*)(Qh + (size_t)(qBase + r) * HD + dq);
        *(uint4*)&sK[0][r][dq] = *(const uint4*)(Kh + (size_t)r * HD + dq);
        alignas(16) bf16_t tv[8];
        *(uint4*)tv = *(const uint4*)(Vh + (size_t)r * HD + dq);
        #pragma unroll
        for (int j = 0; j < 8; ++j) sVt[0][dq + j][r] = tv[j];
    }
    __syncthreads();

    AFragU aq[2];
    #pragma unroll
    for (int ks = 0; ks < 2; ++ks) {
        int r = wave * 16 + ll;
        #pragma unroll
        for (int e = 0; e < 8; ++e) {
            int kk = ks * 32 + ((e < 4) ? 2 * e : 2 * e + 8) + lh * 8;
            aq[ks].u[e] = *(const unsigned*)&sQ[r][kk];
        }
    }

    float m_run[8], l_run[8];
    #pragma unroll
    for (int r = 0; r < 8; ++r) { m_run[r] = -1e30f; l_run[r] = 0.0f; }
    v8f oacc[4] = {};

    const int nKT = T / BKV;
    #pragma unroll 2
    for (int kt = 0; kt < nKT; ++kt) {
        const int cur = kt & 1, nxt = cur ^ 1;
        const bool more = (kt + 1) < nKT;

        uint4 rk[4], rv[4];
        if (more) {
            int kvn = (kt + 1) * BKV;
            #pragma unroll
            for (int i = 0; i < 4; ++i) {
                int idx = tid + i * 128;
                int r = idx >> 3, dq = (idx & 7) * 8;
                rk[i] = *(const uint4*)(Kh + (size_t)(kvn + r) * HD + dq);
                rv[i] = *(const uint4*)(Vh + (size_t)(kvn + r) * HD + dq);
            }
        }

        // ---- S = Q K^T : hoist all 8 B-fragments, then 8 back-to-back WMMAs
        AFragU bk[4][2];
        #pragma unroll
        for (int ni = 0; ni < 4; ++ni) {
            int key = ni * 16 + ll;
            #pragma unroll
            for (int ks = 0; ks < 2; ++ks)
                #pragma unroll
                for (int e = 0; e < 8; ++e) {
                    int kk = ks * 32 + ((e < 4) ? 2 * e : 2 * e + 8) + lh * 8;
                    bk[ni][ks].u[e] = *(const unsigned*)&sK[cur][key][kk];
                }
        }
        v8f sacc[4] = {};
        #pragma unroll
        for (int ks = 0; ks < 2; ++ks)
            #pragma unroll
            for (int ni = 0; ni < 4; ++ni)
                sacc[ni] = WMMA_BF16(aq[ks].v, bk[ni][ks].v, sacc[ni]);

        // ---- online softmax — DPP16 row reductions, no LDS traffic
        #pragma unroll
        for (int r = 0; r < 8; ++r) {
            float mx = fmaxf(fmaxf(sacc[0][r], sacc[1][r]), fmaxf(sacc[2][r], sacc[3][r]));
            mx = rowmax16(mx) * scale;
            float mnew = fmaxf(m_run[r], mx);
            float alpha = __expf(m_run[r] - mnew);
            float rs = 0.0f;
            #pragma unroll
            for (int ni = 0; ni < 4; ++ni) {
                float pexp = __expf(sacc[ni][r] * scale - mnew);
                sacc[ni][r] = pexp;
                rs += pexp;
            }
            rs = rowsum16(rs);
            l_run[r] = l_run[r] * alpha + rs;
            m_run[r] = mnew;
            #pragma unroll
            for (int ni = 0; ni < 4; ++ni) oacc[ni][r] *= alpha;
        }

        // ---- repack P through LDS into A-fragment layout (wave-private rows)
        #pragma unroll
        for (int ni = 0; ni < 4; ++ni)
            #pragma unroll
            for (int r = 0; r < 8; ++r)
                sP[wave * 16 + lh * 8 + r][ni * 16 + ll] = (bf16_t)sacc[ni][r];

        // ---- O += P V : hoist P and V fragments, then 8 back-to-back WMMAs
        AFragU ap[2], bv[4][2];
        {
            int r = wave * 16 + ll;
            #pragma unroll
            for (int ks = 0; ks < 2; ++ks)
                #pragma unroll
                for (int e = 0; e < 8; ++e) {
                    int kk = ks * 32 + ((e < 4) ? 2 * e : 2 * e + 8) + lh * 8;
                    ap[ks].u[e] = *(const unsigned*)&sP[r][kk];
                }
        }
        #pragma unroll
        for (int ni = 0; ni < 4; ++ni) {
            int dim = ni * 16 + ll;
            #pragma unroll
            for (int ks = 0; ks < 2; ++ks)
                #pragma unroll
                for (int e = 0; e < 8; ++e) {
                    int kk = ks * 32 + ((e < 4) ? 2 * e : 2 * e + 8) + lh * 8;
                    bv[ni][ks].u[e] = *(const unsigned*)&sVt[cur][dim][kk];
                }
        }
        #pragma unroll
        for (int ks = 0; ks < 2; ++ks)
            #pragma unroll
            for (int ni = 0; ni < 4; ++ni)
                oacc[ni] = WMMA_BF16(ap[ks].v, bv[ni][ks].v, oacc[ni]);

        if (more) {
            #pragma unroll
            for (int i = 0; i < 4; ++i) {
                int idx = tid + i * 128;
                int r = idx >> 3, dq = (idx & 7) * 8;
                *(uint4*)&sK[nxt][r][dq] = rk[i];
                alignas(16) bf16_t tv[8];
                *(uint4*)tv = rv[i];
                #pragma unroll
                for (int j = 0; j < 8; ++j) sVt[nxt][dq + j][r] = tv[j];
            }
        }
        __syncthreads();
    }

    #pragma unroll
    for (int ni = 0; ni < 4; ++ni)
        #pragma unroll
        for (int r = 0; r < 8; ++r) {
            int t = qBase + wave * 16 + lh * 8 + r;
            int dim = ni * 16 + ll;
            float v = oacc[ni][r] / l_run[r];
            O[((size_t)b * T + t) * D + h * HD + dim] = (bf16_t)v;
        }
}

// ---------------------------------------------------------------------------
// Host orchestration
// ---------------------------------------------------------------------------
extern "C" void kernel_launch(void* const* d_in, const int* in_sizes, int n_in,
                              void* d_out, int out_size, void* d_ws, size_t ws_size,
                              hipStream_t stream) {
    (void)in_sizes; (void)n_in; (void)out_size; (void)ws_size;
    const int*   x         = (const int*)  d_in[0];
    const int*   t_in      = (const int*)  d_in[1];
    const int*   level     = (const int*)  d_in[2];
    const float* tok_emb   = (const float*)d_in[3];
    const float* level_emb = (const float*)d_in[4];
    const float* time_w1   = (const float*)d_in[5];
    const float* time_b1   = (const float*)d_in[6];
    const float* time_w2   = (const float*)d_in[7];
    const float* time_b2   = (const float*)d_in[8];
    const float* ln1_g     = (const float*)d_in[9];
    const float* ln1_b     = (const float*)d_in[10];
    const float* qkv_w     = (const float*)d_in[11];
    const float* proj_w    = (const float*)d_in[12];
    const float* ln2_g     = (const float*)d_in[13];
    const float* ln2_b     = (const float*)d_in[14];
    const float* fc1_w     = (const float*)d_in[15];
    const float* fc2_w     = (const float*)d_in[16];
    const float* lnf_g     = (const float*)d_in[17];
    const float* lnf_b     = (const float*)d_in[18];
    const float* head0_w   = (const float*)d_in[19];

    const int BT = Bc * Tc;   // 8192

    char* p = (char*)d_ws;
    auto alloc = [&](size_t bytes) -> char* {
        char* r = p; p += (bytes + 255) & ~(size_t)255; return r;
    };
    bf16_t* wqkv  = (bf16_t*)alloc((size_t)Lc * Dc * 3 * Dc * sizeof(bf16_t));
    bf16_t* wproj = (bf16_t*)alloc((size_t)Lc * Dc * Dc * sizeof(bf16_t));
    bf16_t* wfc1  = (bf16_t*)alloc((size_t)Lc * Dc * 4 * Dc * sizeof(bf16_t));
    bf16_t* wfc2  = (bf16_t*)alloc((size_t)Lc * 4 * Dc * Dc * sizeof(bf16_t));
    bf16_t* whead = (bf16_t*)alloc((size_t)Dc * V0c * sizeof(bf16_t));
    float*  h     = (float*) alloc((size_t)BT * Dc * sizeof(float));
    bf16_t* hn    = (bf16_t*)alloc((size_t)BT * Dc * sizeof(bf16_t));
    bf16_t* qkvb  = (bf16_t*)alloc((size_t)BT * 3 * Dc * sizeof(bf16_t));
    bf16_t* qb    = (bf16_t*)alloc((size_t)BT * Dc * sizeof(bf16_t));
    bf16_t* kb    = (bf16_t*)alloc((size_t)BT * Dc * sizeof(bf16_t));
    bf16_t* vb    = (bf16_t*)alloc((size_t)BT * Dc * sizeof(bf16_t));
    bf16_t* ob    = (bf16_t*)alloc((size_t)BT * Dc * sizeof(bf16_t));
    bf16_t* a1    = (bf16_t*)alloc((size_t)BT * 4 * Dc * sizeof(bf16_t));
    float*  tefeat= (float*) alloc((size_t)Bc * Dc * sizeof(float));
    float*  teh   = (float*) alloc((size_t)Bc * 4 * Dc * sizeof(float));
    float*  te    = (float*) alloc((size_t)Bc * Dc * sizeof(float));

    auto cdiv = [](long a, long b) { return (int)((a + b - 1) / b); };

    // weight conversion (every call; no cross-call caching allowed)
    f32_to_bf16_kernel<<<4096, 256, 0, stream>>>(qkv_w,  wqkv,  (size_t)Lc * Dc * 3 * Dc);
    f32_to_bf16_kernel<<<2048, 256, 0, stream>>>(proj_w, wproj, (size_t)Lc * Dc * Dc);
    f32_to_bf16_kernel<<<4096, 256, 0, stream>>>(fc1_w,  wfc1,  (size_t)Lc * Dc * 4 * Dc);
    f32_to_bf16_kernel<<<4096, 256, 0, stream>>>(fc2_w,  wfc2,  (size_t)Lc * 4 * Dc * Dc);
    f32_to_bf16_kernel<<<512,  256, 0, stream>>>(head0_w, whead, (size_t)Dc * V0c);

    // time embedding (tiny)
    timefeat_kernel<<<cdiv(Bc * Dc / 2, 256), 256, 0, stream>>>(t_in, tefeat, Bc, Dc);
    small_mm_kernel<<<cdiv((long)Bc * 4 * Dc, 256), 256, 0, stream>>>(
        tefeat, time_w1, time_b1, teh, Bc, 4 * Dc, Dc, 1);
    small_mm_kernel<<<cdiv((long)Bc * Dc, 256), 256, 0, stream>>>(
        teh, time_w2, time_b2, te, Bc, Dc, 4 * Dc, 0);

    // token + level + time embedding
    embed_kernel<<<cdiv((long)BT * Dc, 256), 256, 0, stream>>>(
        x, tok_emb, level_emb, level, te, h, BT, Dc, Tc);

    const dim3 blk(128);
    for (int l = 0; l < Lc; ++l) {
        layernorm_bf16_kernel<<<BT, 256, 0, stream>>>(
            h, ln1_g + (size_t)l * Dc, ln1_b + (size_t)l * Dc, hn, Dc);
        gemm_bf16_kernel<EPI_BF><<<dim3(3 * Dc / 128, BT / 128), blk, 0, stream>>>(
            hn, wqkv + (size_t)l * Dc * 3 * Dc, nullptr, qkvb, nullptr, BT, 3 * Dc, Dc, 0);
        rope_split_kernel<<<cdiv((long)BT * Dc / 2, 256), 256, 0, stream>>>(
            qkvb, qb, kb, vb, Bc, Tc, Hc, HDc);
        attn_kernel<<<dim3(Tc / 64, Bc * Hc), blk, 0, stream>>>(
            qb, kb, vb, ob, Tc, Hc, HDc, 0.125f);
        gemm_bf16_kernel<EPI_RESID><<<dim3(Dc / 128, BT / 128), blk, 0, stream>>>(
            ob, wproj + (size_t)l * Dc * Dc, h, nullptr, h, BT, Dc, Dc, 0);
        layernorm_bf16_kernel<<<BT, 256, 0, stream>>>(
            h, ln2_g + (size_t)l * Dc, ln2_b + (size_t)l * Dc, hn, Dc);
        gemm_bf16_kernel<EPI_GELU_BF><<<dim3(4 * Dc / 128, BT / 128), blk, 0, stream>>>(
            hn, wfc1 + (size_t)l * Dc * 4 * Dc, nullptr, a1, nullptr, BT, 4 * Dc, Dc, 0);
        gemm_bf16_kernel<EPI_RESID><<<dim3(Dc / 128, BT / 128), blk, 0, stream>>>(
            a1, wfc2 + (size_t)l * 4 * Dc * Dc, h, nullptr, h, BT, Dc, 4 * Dc, 0);
    }

    layernorm_bf16_kernel<<<BT, 256, 0, stream>>>(h, lnf_g, lnf_b, hn, Dc);
    gemm_bf16_kernel<EPI_HEAD><<<dim3(cdiv(V0c, 128), BT / 128), blk, 0, stream>>>(
        hn, whead, (float*)d_out, nullptr, nullptr, BT, V0c, Dc, Tc);
}
